// MultiViewAttentionFusion_72164040507835
// MI455X (gfx1250) — matrix operand order repaired and checked by reading, not compiled, not run on hardware
//
#include <hip/hip_runtime.h>

// ---------------- CDNA5 WMMA types ----------------
typedef __bf16 bf16;
typedef __attribute__((ext_vector_type(16))) __bf16 v16bf;
typedef __attribute__((ext_vector_type(8)))  __bf16 v8bf;
typedef __attribute__((ext_vector_type(8)))  float  v8f;
typedef __attribute__((ext_vector_type(4)))  int    v4i;

#ifndef __has_builtin
#define __has_builtin(x) 0
#endif

#if __has_builtin(__builtin_amdgcn_global_load_async_to_lds_b128)
#define USE_ASYNC_COPY 1
#else
#define USE_ASYNC_COPY 0
#endif

#define AS1 __attribute__((address_space(1)))
#define AS3 __attribute__((address_space(3)))

// ---------------- problem constants ----------------
#define C_DIM   64
#define V_DIM   4
#define HW      65536          // H*W = 256*256
#define PX      16             // pixels per block
#define MROWS   64             // PX * V_DIM rows in the per-block GEMM

// LDS strides (elements) with padding
#define AB_S    72             // bf16 A-operand row stride
#define QK_S    200            // bf16 qkv tile row stride (192 cols)
#define AT_S    72             // bf16 attention-out row stride
#define G_S     68             // f32 gate-logits row stride

// LDS layout (bytes); G aliases QKV (QKV dead once attention completes)
#define OFF_P    0                                  // packed x tile [c][v][px] f32
#define OFF_ABF  (C_DIM * V_DIM * PX * 4)           // 16384
#define OFF_QKV  (OFF_ABF + MROWS * AB_S * 2)       // 25600
#define OFF_ATT  (OFF_QKV + MROWS * QK_S * 2)       // 51200
#define SMEM_SZ  (OFF_ATT + MROWS * AT_S * 2)       // 60416 bytes
#define OFF_G    OFF_QKV

__device__ __forceinline__ void wait_async_zero() {
#if __has_builtin(__builtin_amdgcn_s_wait_asynccnt)
    __builtin_amdgcn_s_wait_asynccnt(0);
#else
    asm volatile("s_wait_asynccnt 0x0" ::: "memory");
#endif
}

// ---- A fragment (16x32 bf16) from LDS, contiguous k order (same as B):
// lanes 0-15: elems 0..7 = K k0+0..7,   8..15 = K k0+8..15   (row M = lane)
// lanes16-31: elems 0..7 = K k0+16..23, 8..15 = K k0+24..31  (row M = lane-16)
__device__ __forceinline__ v16bf load_a_lds(const bf16* base, int rs,
                                            int m0, int k0, int lane) {
    const int m   = m0 + (lane & 15);
    const int klo = k0 + ((lane & 16) ? 16 : 0);
    const int khi = k0 + ((lane & 16) ? 24 : 8);
    v8bf lo = *(const v8bf*)(base + m * rs + klo);
    v8bf hi = *(const v8bf*)(base + m * rs + khi);
    v16bf a;
#pragma unroll
    for (int e = 0; e < 8; ++e) { a[e] = lo[e]; a[8 + e] = hi[e]; }
    return a;
}

// ---- B fragment (32x16 bf16) straight from global f32 weights W[n][64]
// (computes X @ W^T, so B[k][n] = W[n][k]).  contiguous k order (same as A):
// lanes 0-15: elems 0..15 = K k0+0..15  (col N = lane)
// lanes16-31: elems 0..15 = K k0+16..31 (col N = lane-16)
__device__ __forceinline__ v16bf load_b_gl(const float* __restrict__ w,
                                           int n0, int k0, int lane) {
    const int nn = n0 + (lane & 15);
    const int kb = k0 + ((lane & 16) ? 16 : 0);
    const float4* p = (const float4*)(w + nn * 64 + kb);
    v16bf b;
#pragma unroll
    for (int q = 0; q < 4; ++q) {
        float4 f = p[q];
        b[q * 4 + 0] = (bf16)f.x; b[q * 4 + 1] = (bf16)f.y;
        b[q * 4 + 2] = (bf16)f.z; b[q * 4 + 3] = (bf16)f.w;
    }
    return b;
}

__global__ __launch_bounds__(256)
void mva_fused_kernel(const float* __restrict__ x,
                      const float* __restrict__ wqkv,
                      const float* __restrict__ wout,
                      float* __restrict__ out) {
    __shared__ __align__(16) char smem[SMEM_SZ];
    float* P   = (float*)(smem + OFF_P);    // packed x tile [c][v][px] f32
    bf16*  ABF = (bf16*)(smem + OFF_ABF);
    bf16*  QKV = (bf16*)(smem + OFF_QKV);
    bf16*  ATT = (bf16*)(smem + OFF_ATT);
    float* G   = (float*)(smem + OFF_G);    // aliases QKV region

    const int t    = threadIdx.x;
    const int lane = t & 31;
    const int wave = t >> 5;

    const int p0  = blockIdx.x * PX;        // 16 consecutive pixels along W
    const int n   = p0 >> 16;               // p0 / (H*W)
    const int hh  = (p0 >> 8) & 255;
    const int ww0 = p0 & 255;
    const long inbase = (long)n * ((long)C_DIM * V_DIM * HW) + hh * 256 + ww0;

    // -------- phase 0: warm the weight matrices (global_prefetch_b8) --------
    if (t < 192) __builtin_prefetch(wqkv + t * 64, 0, 3);
    else         __builtin_prefetch(wout + (t - 192) * 64, 0, 3);

    // -------- phase 1: x tile -> LDS, packed [c][v][px] (16B per quad) ------
    // quad qi = c*16 + v*4 + pq maps 1:1 between the global float4 at
    // (c,v,pixels pq*4..pq*4+3) and LDS bytes [qi*16, qi*16+16).
#if USE_ASYNC_COPY
#pragma unroll
    for (int it = 0; it < 4; ++it) {
        const int qi = t + it * 256;
        const int c  = qi >> 4;
        const int v  = (qi >> 2) & 3;
        const int pq = qi & 3;
        const float* gsrc = x + inbase + (long)(c * 4 + v) * HW + pq * 4;
        __builtin_amdgcn_global_load_async_to_lds_b128(
            (AS1 v4i*)(unsigned long long)(size_t)gsrc,
            (AS3 v4i*)(unsigned int)(size_t)(smem + OFF_P + qi * 16),
            0, 0);
    }
    wait_async_zero();          // this wave's transfers complete
#else
#pragma unroll
    for (int it = 0; it < 4; ++it) {
        const int qi = t + it * 256;
        const int c  = qi >> 4;
        const int v  = (qi >> 2) & 3;
        const int pq = qi & 3;
        const float4 val = *(const float4*)(x + inbase + (long)(c * 4 + v) * HW + pq * 4);
        *(float4*)(P + qi * 4) = val;
    }
#endif

    // -------- phase 1b: unpack own quads into bf16 A-operand layout ---------
#pragma unroll
    for (int it = 0; it < 4; ++it) {
        const int qi = t + it * 256;
        const int c  = qi >> 4;
        const int v  = (qi >> 2) & 3;
        const int pq = qi & 3;
        const float4 val = *(const float4*)(P + qi * 4);
        const float f[4] = {val.x, val.y, val.z, val.w};
#pragma unroll
        for (int u = 0; u < 4; ++u) {
            const int row = (pq * 4 + u) * V_DIM + v;   // (pixel, view) row
            ABF[row * AB_S + c] = (bf16)f[u];
        }
    }
    __syncthreads();

    // ---------------- phase 2: QKV GEMM  (64x64) @ (64x192) -----------------
#pragma unroll
    for (int i = 0; i < 6; ++i) {
        const int tile = wave + i * 8;
        const int m0 = (tile / 12) * 16;
        const int n0 = (tile % 12) * 16;
        v8f acc = {};
#pragma unroll
        for (int k0 = 0; k0 < 64; k0 += 32) {
            v16bf a = load_a_lds(ABF, AB_S, m0, k0, lane);
            v16bf b = load_b_gl(wqkv, n0, k0, lane);
            acc = __builtin_amdgcn_wmma_f32_16x16x32_bf16(
                false, a, false, b, (short)0, acc, false, false);
        }
        const int nn   = n0 + (lane & 15);
        const int roff = (lane & 16) ? 8 : 0;
#pragma unroll
        for (int r = 0; r < 8; ++r)
            QKV[(m0 + roff + r) * QK_S + nn] = (bf16)acc[r];
    }
    __syncthreads();

    // ---------------- phase 3: per-pixel 2-head attention over V=4 -----------
    if (t < 128) {                          // (pixel, head, query-row)
        const int px   = t >> 3;
        const int head = (t >> 2) & 1;
        const int i    = t & 3;
        const int r0   = px * V_DIM;
        const int qo   = head * 32;
        const int ko   = 64 + head * 32;
        const int vo   = 128 + head * 32;
        const float scale = 0.1767766952966369f;   // 32^-0.5

        float sim[4];
#pragma unroll
        for (int j = 0; j < 4; ++j) {
            float s = 0.f;
#pragma unroll
            for (int d = 0; d < 32; ++d)
                s += (float)QKV[(r0 + i) * QK_S + qo + d] *
                     (float)QKV[(r0 + j) * QK_S + ko + d];
            sim[j] = s * scale;
        }
        const float m = fmaxf(fmaxf(sim[0], sim[1]), fmaxf(sim[2], sim[3]));
        float p[4], sum = 0.f;
#pragma unroll
        for (int j = 0; j < 4; ++j) { p[j] = __expf(sim[j] - m); sum += p[j]; }
        const float inv = 1.f / sum;
#pragma unroll
        for (int j = 0; j < 4; ++j) p[j] *= inv;
#pragma unroll
        for (int d = 0; d < 32; ++d) {
            float o = 0.f;
#pragma unroll
            for (int j = 0; j < 4; ++j)
                o += p[j] * (float)QKV[(r0 + j) * QK_S + vo + d];
            ATT[(r0 + i) * AT_S + qo + d] = (bf16)o;
        }
    }
    __syncthreads();

    // ---------------- phase 4: gate GEMM  (64x64) @ (64x64) ------------------
#pragma unroll
    for (int i = 0; i < 2; ++i) {
        const int tile = wave + i * 8;
        const int m0 = (tile >> 2) * 16;
        const int n0 = (tile & 3) * 16;
        v8f acc = {};
#pragma unroll
        for (int k0 = 0; k0 < 64; k0 += 32) {
            v16bf a = load_a_lds(ATT, AT_S, m0, k0, lane);
            v16bf b = load_b_gl(wout, n0, k0, lane);
            acc = __builtin_amdgcn_wmma_f32_16x16x32_bf16(
                false, a, false, b, (short)0, acc, false, false);
        }
        const int nn   = n0 + (lane & 15);
        const int roff = (lane & 16) ? 8 : 0;
#pragma unroll
        for (int r = 0; r < 8; ++r)
            G[(m0 + roff + r) * G_S + nn] = acc[r];
    }
    __syncthreads();

    // ------- phase 5: softmax over views, gate*xf, sum over views, store -----
    // thread -> (channel c, pixel quad pq); one float4 store per thread.
    // xf[px][v][c] == P[c*64 + v*16 + px].
    const long outbase = (long)n * ((long)C_DIM * HW) + hh * 256 + ww0;
    {
        const int c  = t >> 2;
        const int pq = t & 3;
        float r[4];
#pragma unroll
        for (int u = 0; u < 4; ++u) {
            const int px = pq * 4 + u;
            const int rb = px * V_DIM;
            const float g0 = G[(rb + 0) * G_S + c];
            const float g1 = G[(rb + 1) * G_S + c];
            const float g2 = G[(rb + 2) * G_S + c];
            const float g3 = G[(rb + 3) * G_S + c];
            const float m  = fmaxf(fmaxf(g0, g1), fmaxf(g2, g3));
            const float e0 = __expf(g0 - m), e1 = __expf(g1 - m);
            const float e2 = __expf(g2 - m), e3 = __expf(g3 - m);
            const float inv = 1.f / (e0 + e1 + e2 + e3);
            const int pb = c * 64 + px;
            r[u] = (e0 * P[pb +  0] + e1 * P[pb + 16] +
                    e2 * P[pb + 32] + e3 * P[pb + 48]) * inv;
        }
        float4 res; res.x = r[0]; res.y = r[1]; res.z = r[2]; res.w = r[3];
        *(float4*)(out + outbase + (long)c * HW + pq * 4) = res;
    }
}

extern "C" void kernel_launch(void* const* d_in, const int* in_sizes, int n_in,
                              void* d_out, int out_size, void* d_ws, size_t ws_size,
                              hipStream_t stream) {
    const float* x     = (const float*)d_in[0];   // (N, 64, 4, 256, 256) f32
    const float* wqkv  = (const float*)d_in[1];   // (192, 64) f32
    const float* wout  = (const float*)d_in[2];   // (64, 64) f32
    float*       out   = (float*)d_out;           // (N, 64, 256, 256) f32

    const int npix   = in_sizes[0] / (C_DIM * V_DIM);   // N*H*W
    const int blocks = npix / PX;

    mva_fused_kernel<<<blocks, 256, 0, stream>>>(x, wqkv, wout, out);
}
